// MMN_7361573945989
// MI455X (gfx1250) — matrix operand order repaired and checked by reading, not compiled, not run on hardware
//
#include <hip/hip_runtime.h>
#include <hip/hip_bf16.h>

typedef __attribute__((ext_vector_type(16))) __bf16 v16bf;
typedef __attribute__((ext_vector_type(8)))  float  v8f;

#define HW 3600
#define PW 3712            // padded pixel dim = 29 * 128
#define KTOT 3072
#define KP_TOT 1536        // bf16 pairs
#define CV 512

union bfpair { __bf16 h[2]; unsigned int u; };

// --- CDNA5 async global->LDS copy (ASYNCcnt path), with safe fallback -------
__device__ __forceinline__ void async_copy_dword(const unsigned int* g,
                                                 unsigned int* l) {
#if __has_builtin(__builtin_amdgcn_global_load_async_to_lds_b32)
  __builtin_amdgcn_global_load_async_to_lds_b32(
      (__attribute__((address_space(1))) int*)g,
      (__attribute__((address_space(3))) int*)l, 0, 0);
#else
  *l = *g;
#endif
}
__device__ __forceinline__ void async_wait_all() {
#if __has_builtin(__builtin_amdgcn_s_wait_asynccnt)
  __builtin_amdgcn_s_wait_asynccnt(0);
#endif
}

__device__ __forceinline__ v8f wmma_bf16(const v16bf& a, const v16bf& b,
                                         const v8f& c) {
  return __builtin_amdgcn_wmma_f32_16x16x32_bf16(false, a, false, b, (short)0,
                                                 c, false, false);
}

// ---------------------------------------------------------------------------
// K1: per-pixel L2 normalize over channels, fold (TEMP * w_red[widx]) into the
// query side, emit bf16 PAIRS: dword layout [b][kpair][PW]; pixels >= HW are
// zero-filled so the GEMMs need no bounds checks at all.
// ---------------------------------------------------------------------------
__global__ __launch_bounds__(256) void pack_norm(const float* __restrict__ src,
                                                 unsigned int* __restrict__ dst,
                                                 int C, int kpoff,
                                                 const float* __restrict__ wred,
                                                 int widx) {
  int p = blockIdx.x * 256 + threadIdx.x;
  int b = blockIdx.y;
  if (p >= PW) return;
  unsigned int* d = dst + ((size_t)b * KP_TOT + kpoff) * PW;
  if (p >= HW) {
    for (int kp = 0; kp < C / 2; ++kp) d[(size_t)kp * PW + p] = 0u;
    return;
  }
  const float* s = src + (size_t)b * C * HW;
  float ss = 0.f;
  for (int c = 0; c < C; ++c) {
    float x = s[(size_t)c * HW + p];
    ss += x * x;
  }
  float scale = (widx >= 0) ? 20.0f * wred[widx] : 1.0f;   // TEMP = 20
  float inv = scale / fmaxf(sqrtf(ss), 1e-12f);
  for (int kp = 0; kp < C / 2; ++kp) {
    bfpair t;
    t.h[0] = (__bf16)(s[(size_t)(2 * kp) * HW + p] * inv);
    t.h[1] = (__bf16)(s[(size_t)(2 * kp + 1) * HW + p] * inv);
    d[(size_t)kp * PW + p] = t.u;
  }
}

// ---------------------------------------------------------------------------
// K2: red[p,q] = sum_k Q[k,p]*S[k,q]  (padded 3712x3712, K=3072).
// 256 threads (8 waves), block tile 128x128, wave tile 32x64.  Double-buffered
// LDS: async global->LDS copies for K-step i+1 overlap the 8 WMMAs of step i.
// No guards anywhere (padded inputs/outputs; pad region computes exact zeros).
// ---------------------------------------------------------------------------
__global__ __launch_bounds__(256) void red_gemm(const unsigned int* __restrict__ Q2,
                                                const unsigned int* __restrict__ S2,
                                                float* __restrict__ red, int b) {
  __shared__ unsigned int As[2][128][18];   // [buf][m][kpair], pad->no conflicts
  __shared__ unsigned int Bs[2][128][18];   // [buf][n][kpair]
  const int tid  = threadIdx.x;
  const int wave = tid >> 5, lane = tid & 31;
  const int ml = lane & 15, hi = lane >> 4;
  const int wrow = wave >> 1, wcol = wave & 1;
  const int p0 = blockIdx.y * 128, q0 = blockIdx.x * 128;
  const unsigned int* Qb = Q2 + (size_t)b * KP_TOT * PW;
  const unsigned int* Sb = S2 + (size_t)b * KP_TOT * PW;

  const int kpI = tid >> 7;   // 0/1: kpair row within a 2-row chunk
  const int mm  = tid & 127;  // pixel within tile
  const unsigned offA = (unsigned)(kpI * PW + p0 + mm);
  const unsigned offB = (unsigned)(kpI * PW + q0 + mm);

  v8f acc[2][4];
#pragma unroll
  for (int s = 0; s < 2; ++s)
#pragma unroll
    for (int n = 0; n < 4; ++n) acc[s][n] = {};

  // prologue: stage K-step 0 into buffer 0
#pragma unroll
  for (int i = 0; i < 8; ++i) {
    async_copy_dword(Qb + offA + (unsigned)(i * 2 * PW), &As[0][mm][kpI + 2 * i]);
    async_copy_dword(Sb + offB + (unsigned)(i * 2 * PW), &Bs[0][mm][kpI + 2 * i]);
  }
  async_wait_all();
  __syncthreads();

  int cur = 0;
  for (int kp0 = 0; kp0 < KP_TOT; kp0 += 16) {
    const int nxt = cur ^ 1;
    if (kp0 + 16 < KP_TOT) {          // issue async stage of next K-step
      unsigned base = (unsigned)((kp0 + 16) * PW);
#pragma unroll
      for (int i = 0; i < 8; ++i) {
        async_copy_dword(Qb + base + offA + (unsigned)(i * 2 * PW),
                         &As[nxt][mm][kpI + 2 * i]);
        async_copy_dword(Sb + base + offB + (unsigned)(i * 2 * PW),
                         &Bs[nxt][mm][kpI + 2 * i]);
      }
    }
    // load all fragments, then burst 8 WMMAs
    v16bf af[2], bf[4];
#pragma unroll
    for (int s = 0; s < 2; ++s) {
      const unsigned int* row = &As[cur][wrow * 16 + s * 64 + ml][0];
#pragma unroll
      for (int v = 0; v < 8; ++v)
        ((unsigned int*)&af[s])[v] = row[((v >> 2) << 3) + (hi << 2) + (v & 3)];
    }
#pragma unroll
    for (int nf = 0; nf < 4; ++nf) {
      const unsigned int* row = &Bs[cur][wcol * 64 + nf * 16 + ml][0];
#pragma unroll
      for (int v = 0; v < 8; ++v)
        ((unsigned int*)&bf[nf])[v] = row[((v >> 2) << 3) + (hi << 2) + (v & 3)];
    }
#pragma unroll
    for (int nf = 0; nf < 4; ++nf) {
      acc[0][nf] = wmma_bf16(af[0], bf[nf], acc[0][nf]);
      acc[1][nf] = wmma_bf16(af[1], bf[nf], acc[1][nf]);
    }
    async_wait_all();
    __syncthreads();
    cur = nxt;
  }

#pragma unroll
  for (int s = 0; s < 2; ++s)
#pragma unroll
    for (int nf = 0; nf < 4; ++nf)
#pragma unroll
      for (int r = 0; r < 8; ++r) {
        int row = p0 + wrow * 16 + s * 64 + r + 8 * hi;
        int col = q0 + wcol * 64 + nf * 16 + ml;
        red[(size_t)row * PW + col] = acc[s][nf][r];   // padded, unguarded
      }
}

// ---------------------------------------------------------------------------
// K3: in-place row softmax over the 3600 valid support pixels (stride PW).
// Pad columns stay exactly zero -> zero attention weight downstream.
// ---------------------------------------------------------------------------
__global__ __launch_bounds__(256) void softmax_rows(float* __restrict__ red) {
  __shared__ float sdata[256];
  float* r = red + (size_t)blockIdx.x * PW;
  int tid = threadIdx.x;
  float m = -1e30f;
  for (int j = tid; j < HW; j += 256) m = fmaxf(m, r[j]);
  sdata[tid] = m;
  __syncthreads();
  for (int s = 128; s > 0; s >>= 1) {
    if (tid < s) sdata[tid] = fmaxf(sdata[tid], sdata[tid + s]);
    __syncthreads();
  }
  m = sdata[0];
  __syncthreads();
  float sum = 0.f;
  for (int j = tid; j < HW; j += 256) {
    float e = __expf(r[j] - m);
    r[j] = e;
    sum += e;
  }
  sdata[tid] = sum;
  __syncthreads();
  for (int s = 128; s > 0; s >>= 1) {
    if (tid < s) sdata[tid] += sdata[tid + s];
    __syncthreads();
  }
  float inv = 1.0f / sdata[0];
  for (int j = tid; j < HW; j += 256) r[j] *= inv;
}

// ---------------------------------------------------------------------------
// K4: att_fq[c,p] = sum_q attn[p,q] * f_s[c,q].  M=3712(p,pad), N=512, K=3600.
// A staged unguarded from padded attn; f_s staging uses a uniform full/partial
// branch (only the last K-iteration is guarded).
// ---------------------------------------------------------------------------
__global__ __launch_bounds__(256) void attend(const float* __restrict__ attn,
                                              const float* __restrict__ fs,
                                              float* __restrict__ out, int b) {
  __shared__ unsigned int As[128][18];   // [m=p][kpair=q/2]
  __shared__ unsigned int Bs[128][18];   // [n=c][kpair=q/2]
  const int tid  = threadIdx.x;
  const int wave = tid >> 5, lane = tid & 31;
  const int ml = lane & 15, hi = lane >> 4;
  const int wrow = wave >> 1, wcol = wave & 1;
  const int p0 = blockIdx.x * 128, c0 = blockIdx.y * 128;
  const float* fsb = fs + (size_t)b * CV * HW;
  float* ob = out + (size_t)b * CV * HW;

  const int rr = tid >> 4;            // 0..15 row-in-chunk
  const int kpB = tid & 15;           // kpair
  const unsigned aOff = (unsigned)((p0 + rr) * PW + 2 * kpB);

  v8f acc[2][4];
#pragma unroll
  for (int s = 0; s < 2; ++s)
#pragma unroll
    for (int n = 0; n < 4; ++n) acc[s][n] = {};

  for (int q0 = 0; q0 < HW; q0 += 32) {
    // A: 128 rows(p) x 16 kpairs from padded attn (fp32 -> bf16 pair)
#pragma unroll
    for (int i = 0; i < 8; ++i) {
      float2 x = *(const float2*)(attn + aOff + (unsigned)q0 +
                                  (unsigned)(i * 16 * PW));
      bfpair t;
      t.h[0] = (__bf16)x.x;
      t.h[1] = (__bf16)x.y;
      As[i * 16 + rr][kpB] = t.u;
    }
    // B: 128 rows(c) x 16 kpairs from f_s
    if (q0 + 32 <= HW) {
#pragma unroll
      for (int i = 0; i < 8; ++i) {
        float2 x = *(const float2*)(fsb + (size_t)(c0 + rr + 16 * i) * HW +
                                    q0 + 2 * kpB);
        bfpair t;
        t.h[0] = (__bf16)x.x;
        t.h[1] = (__bf16)x.y;
        Bs[rr + 16 * i][kpB] = t.u;
      }
    } else {
#pragma unroll
      for (int i = 0; i < 8; ++i) {
        int q = q0 + 2 * kpB;
        bfpair t;
        if (q + 1 < HW) {
          float2 x = *(const float2*)(fsb + (size_t)(c0 + rr + 16 * i) * HW + q);
          t.h[0] = (__bf16)x.x;
          t.h[1] = (__bf16)x.y;
        } else {
          t.u = 0u;
        }
        Bs[rr + 16 * i][kpB] = t.u;
      }
    }
    __syncthreads();

    v16bf af[2], bf[4];
#pragma unroll
    for (int s = 0; s < 2; ++s) {
      const unsigned int* row = &As[wrow * 16 + s * 64 + ml][0];
#pragma unroll
      for (int v = 0; v < 8; ++v)
        ((unsigned int*)&af[s])[v] = row[((v >> 2) << 3) + (hi << 2) + (v & 3)];
    }
#pragma unroll
    for (int nf = 0; nf < 4; ++nf) {
      const unsigned int* row = &Bs[wcol * 64 + nf * 16 + ml][0];
#pragma unroll
      for (int v = 0; v < 8; ++v)
        ((unsigned int*)&bf[nf])[v] = row[((v >> 2) << 3) + (hi << 2) + (v & 3)];
    }
#pragma unroll
    for (int nf = 0; nf < 4; ++nf) {
      acc[0][nf] = wmma_bf16(af[0], bf[nf], acc[0][nf]);
      acc[1][nf] = wmma_bf16(af[1], bf[nf], acc[1][nf]);
    }
    __syncthreads();
  }

#pragma unroll
  for (int s = 0; s < 2; ++s)
#pragma unroll
    for (int nf = 0; nf < 4; ++nf)
#pragma unroll
      for (int r = 0; r < 8; ++r) {
        int p = p0 + wrow * 16 + s * 64 + r + 8 * hi;
        int c = c0 + wcol * 64 + nf * 16 + ml;
        if (p < HW) ob[(size_t)c * HW + p] = acc[s][nf][r];
      }
}

// ---------------------------------------------------------------------------
// K5: fq = l2norm(f_q) + l2norm(att_fq) * 0.3   (per pixel over 512 channels)
// ---------------------------------------------------------------------------
__global__ __launch_bounds__(256) void final_norm(const float* __restrict__ fq,
                                                  const float* __restrict__ att,
                                                  float* __restrict__ out) {
  int p = blockIdx.x * 256 + threadIdx.x;
  int b = blockIdx.y;
  if (p >= HW) return;
  size_t base = (size_t)b * CV * HW;
  float s1 = 0.f, s2 = 0.f;
  for (int c = 0; c < CV; ++c) {
    float x = fq[base + (size_t)c * HW + p];
    float y = att[base + (size_t)c * HW + p];
    s1 += x * x;
    s2 += y * y;
  }
  float i1 = 1.0f / fmaxf(sqrtf(s1), 1e-12f);
  float i2 = 0.3f / fmaxf(sqrtf(s2), 1e-12f);
  for (int c = 0; c < CV; ++c) {
    size_t o = base + (size_t)c * HW + p;
    out[o] = fq[o] * i1 + att[o] * i2;
  }
}

extern "C" void kernel_launch(void* const* d_in, const int* in_sizes, int n_in,
                              void* d_out, int out_size, void* d_ws, size_t ws_size,
                              hipStream_t stream) {
  const float* fq_l3 = (const float*)d_in[0];
  const float* fs_l3 = (const float*)d_in[1];
  const float* fq_l4 = (const float*)d_in[2];
  const float* fs_l4 = (const float*)d_in[3];
  const float* f_q   = (const float*)d_in[4];
  const float* f_s   = (const float*)d_in[5];
  const float* w_red = (const float*)d_in[6];

  // workspace layout (146.3 MB total), PW-padded
  unsigned int* Q2 = (unsigned int*)d_ws;                        // 45,613,056 B
  unsigned int* S2 = (unsigned int*)((char*)d_ws + 45613056);    // 45,613,056 B
  float*       red = (float*)((char*)d_ws + 91226112);           // 55,115,776 B

  float* out_fq  = (float*)d_out;
  float* out_att = out_fq + (size_t)2 * CV * HW;

  dim3 pb(256), pg((PW + 255) / 256, 2);
  // channel concat order: layer4 first (w_red[0]), then layer3 (w_red[1])
  pack_norm<<<pg, pb, 0, stream>>>(fq_l4, Q2, 2048, 0,    w_red, 0);
  pack_norm<<<pg, pb, 0, stream>>>(fq_l3, Q2, 1024, 1024, w_red, 1);
  pack_norm<<<pg, pb, 0, stream>>>(fs_l4, S2, 2048, 0,    w_red, -1);
  pack_norm<<<pg, pb, 0, stream>>>(fs_l3, S2, 1024, 1024, w_red, -1);

  const int T = PW / 128;   // 29 tiles, exact
  for (int b = 0; b < 2; ++b) {
    red_gemm<<<dim3(T, T), 256, 0, stream>>>(Q2, S2, red, b);
    softmax_rows<<<HW, 256, 0, stream>>>(red);
    attend<<<dim3(T, CV / 128), 256, 0, stream>>>(red, f_s, out_att, b);
  }
  final_norm<<<dim3((HW + 255) / 256, 2), pb, 0, stream>>>(f_q, out_att, out_fq);
}
